// MultiScaleSA_6734508720336
// MI455X (gfx1250) — matrix operand order repaired and checked by reading, not compiled, not run on hardware
//
#include <hip/hip_runtime.h>
#include <hip/hip_bf16.h>

typedef _Float16 v8h  __attribute__((ext_vector_type(8)));
typedef _Float16 v16h __attribute__((ext_vector_type(16)));
typedef float    v8f  __attribute__((ext_vector_type(8)));
typedef unsigned int u32x4 __attribute__((ext_vector_type(4)));
typedef int          i32x8 __attribute__((ext_vector_type(8)));
typedef int          i32x4 __attribute__((ext_vector_type(4)));

#define NBATCH 16
#define NP_    4096
#define MP_    1024
#define NC_    (NBATCH * MP_)   // 16384 centroids
#define INC_   32
#define TOPK_  64
#define OUTC_  320
#define MAX_W_HALVES 12288      // max Ktiles*NT*512 (K=96, Cout=128) = 24KB

static __device__ __forceinline__ v16h combine16(v8h lo, v8h hi) {
  return __builtin_shufflevector(lo, hi, 0,1,2,3,4,5,6,7,8,9,10,11,12,13,14,15);
}

// ---------------------------------------------------------------- zero stats
__global__ void zero_kernel(float* p, int n) {
  int i = blockIdx.x * blockDim.x + threadIdx.x;
  if (i < n) p[i] = 0.0f;
}

// ---------------------------------------------------- top-64 nearest search
__global__ __launch_bounds__(256)
void topk_kernel(const float* __restrict__ pos, const int* __restrict__ idx,
                 int* __restrict__ nbr_idx, float* __restrict__ nbr_d2) {
  __shared__ float d2s[NP_];
  __shared__ float rv[256];
  __shared__ int   ra[256];
  int c = blockIdx.x;            // centroid id
  int b = c / MP_;               // cloud id
  int t = threadIdx.x;
  int ctr = idx[c];
  float cx = pos[ctr * 3 + 0], cy = pos[ctr * 3 + 1], cz = pos[ctr * 3 + 2];
  const float* pb = pos + (size_t)b * NP_ * 3;
  for (int p = t; p < NP_; p += 256) {
    float dx = pb[p * 3 + 0] - cx;
    float dy = pb[p * 3 + 1] - cy;
    float dz = pb[p * 3 + 2] - cz;
    d2s[p] = dx * dx + dy * dy + dz * dz;
  }
  __syncthreads();
  for (int r = 0; r < TOPK_; ++r) {
    float best = __builtin_huge_valf(); int barg = 0;
    for (int p = t; p < NP_; p += 256) {
      float v = d2s[p];
      if (v < best) { best = v; barg = p; }
    }
    rv[t] = best; ra[t] = barg;
    __syncthreads();
    for (int s = 128; s > 0; s >>= 1) {
      if (t < s && rv[t + s] < rv[t]) { rv[t] = rv[t + s]; ra[t] = ra[t + s]; }
      __syncthreads();
    }
    if (t == 0) {
      int a = ra[0];
      nbr_idx[(size_t)c * TOPK_ + r] = b * NP_ + a;
      nbr_d2[(size_t)c * TOPK_ + r] = rv[0];
      d2s[a] = __builtin_huge_valf();
    }
    __syncthreads();
  }
}

// -------------------------------------------------- pack weights (B frags)
// [ktile][ntile][lane 0..31][16 halves]; lane<16: K=t*32+{0..7,16..23} of column
// nt*16+lane; lane>=16: K=t*32+{8..15,24..31} (ISA 16-bit B-matrix layout).
__global__ void pack_w_kernel(const float* __restrict__ W, int cin, int cout,
                              int ntiles, _Float16* __restrict__ dst, int elems) {
  int i = blockIdx.x * blockDim.x + threadIdx.x;
  if (i >= elems) return;
  int h    = i & 15;
  int lane = (i >> 4) & 31;
  int nt   = (i >> 9) % ntiles;
  int t    = (i >> 9) / ntiles;
  int col  = nt * 16 + (lane & 15);
  int k    = t * 32 + ((lane >> 4) * 8) + (h & 7) + ((h >> 3) * 16);
  float v = (k < cin) ? W[(size_t)k * cout + col] : 0.0f;
  dst[i] = (_Float16)v;
}

// ----------------------------------------------------- gather edge features
__global__ __launch_bounds__(256)
void pack_edges_kernel(const float* __restrict__ x, const float* __restrict__ pos,
                       const int* __restrict__ idx, const int* __restrict__ nbr_idx,
                       const float* __restrict__ nbr_d2, unsigned char* __restrict__ mask,
                       float* __restrict__ cnt, _Float16* __restrict__ dst,
                       int k, float r2, int E) {
  __shared__ int scnt[256];
  int e = blockIdx.x * 256 + threadIdx.x;
  int valid = 0;
  if (e < E) {
    int c = e / k, j = e - c * k;
    int nb = nbr_idx[(size_t)c * TOPK_ + j];
    float d2 = nbr_d2[(size_t)c * TOPK_ + j];
    valid = (d2 <= r2) ? 1 : 0;
    mask[e] = (unsigned char)valid;
    int ctr = idx[c];
    float rx = pos[nb * 3 + 0] - pos[ctr * 3 + 0];
    float ry = pos[nb * 3 + 1] - pos[ctr * 3 + 1];
    float rz = pos[nb * 3 + 2] - pos[ctr * 3 + 2];
    _Float16* d = dst + (size_t)e * 64;
    const float* xs = x + (size_t)nb * INC_;
#pragma unroll
    for (int q = 0; q < INC_; ++q) d[q] = (_Float16)xs[q];
    d[32] = (_Float16)rx; d[33] = (_Float16)ry; d[34] = (_Float16)rz;
#pragma unroll
    for (int q = 35; q < 64; ++q) d[q] = (_Float16)0.0f;
  }
  scnt[threadIdx.x] = valid;
  __syncthreads();
  for (int s = 128; s > 0; s >>= 1) {
    if (threadIdx.x < s) scnt[threadIdx.x] += scnt[threadIdx.x + s];
    __syncthreads();
  }
  if (threadIdx.x == 0) atomicAdd(cnt, (float)scnt[0]);
}

// --------------------------------------------------------------- WMMA GEMM
// 4 waves/block, each wave owns one 16-row M tile and ALL NT output tiles
// (A read exactly once). Packed W staged block-wide in LDS via the Tensor
// Data Mover (wave 0 issues a 1-D tile D#); B fragments come back as
// ds_load_b128.
template <int NT>
__global__ __launch_bounds__(128)
void gemm_kernel(const _Float16* __restrict__ A, const _Float16* __restrict__ Wp,
                 const float* __restrict__ bias, _Float16* __restrict__ H,
                 int Ktiles) {
  constexpr int Cout = NT * 16;
  __shared__ _Float16 sW[MAX_W_HALVES];
  int tid  = threadIdx.x;
  int lane = tid & 31;
  int wave = tid >> 5;
  int nhalves = Ktiles * NT * 512;

#if __has_builtin(__builtin_amdgcn_tensor_load_to_lds)
  if (wave == 0) {
    unsigned ldsa = (unsigned)(uintptr_t)&sW[0];            // addr[31:0] = LDS
    unsigned long long ga = (unsigned long long)(uintptr_t)Wp;
    unsigned ndw = (unsigned)(nhalves >> 1);                // payload in DWORDs
    u32x4 g0 = {};
    g0[0] = 1u;                                             // count=1 (valid D#)
    g0[1] = ldsa;                                           // lds_addr
    g0[2] = (unsigned)(ga & 0xFFFFFFFFu);                   // global_addr[31:0]
    g0[3] = (unsigned)((ga >> 32) & 0x1FFFFFFu) | (2u << 30); // addr hi | type=2
    i32x8 g1 = {};
    g1[0] = (int)(2u << 16);                                // wg_mask=0, 4B elems
    g1[1] = (int)((ndw & 0xFFFFu) << 16);                   // tensor_dim0[15:0]
    g1[2] = (int)((ndw >> 16) | (1u << 16));                // dim0 hi | dim1=1
    g1[3] = (int)((ndw & 0xFFFFu) << 16);                   // tile_dim0 = ndw
    g1[4] = 1;                                              // tile_dim1=1
    g1[5] = (int)ndw;                                       // dim0_stride lo
    g1[6] = (int)((ndw & 0xFFFFu) << 16);                   // dim1_stride lo16
    g1[7] = (int)(ndw >> 16);
    i32x4 gz4 = {};
    i32x8 gz8 = {};
    __builtin_amdgcn_tensor_load_to_lds(g0, g1, gz4, gz4, gz8, 0);
    __builtin_amdgcn_s_wait_tensorcnt(0);
  }
  __syncthreads();
#else
  for (int i = tid * 8; i < nhalves; i += 128 * 8)
    *(v8h*)&sW[i] = *(const v8h*)&Wp[i];
  __syncthreads();
#endif

  int mt   = blockIdx.x * 4 + wave;
  int Kpad = Ktiles * 32;
  int row0 = mt * 16 + (lane & 15);
  int ko   = (lane >> 4) * 8;
  v8f acc[NT] = {};
  for (int t = 0; t < Ktiles; ++t) {
    const _Float16* ap = A + (size_t)row0 * Kpad + t * 32 + ko;
    __builtin_prefetch((const void*)(ap + 32), 0, 0);       // next K tile
    v16h a = combine16(*(const v8h*)ap, *(const v8h*)(ap + 16));
#pragma unroll
    for (int nt = 0; nt < NT; ++nt) {
      const _Float16* bp = &sW[(size_t)(((t * NT + nt) * 32) + lane) * 16];
      v16h bfr = combine16(*(const v8h*)bp, *(const v8h*)(bp + 8));
      acc[nt] = __builtin_amdgcn_wmma_f32_16x16x32_f16(
          false, a, false, bfr, (short)0, acc[nt], false, false);
    }
  }
  int colb = lane & 15;
  int m0   = mt * 16 + ((lane >> 4) * 8);
#pragma unroll
  for (int nt = 0; nt < NT; ++nt) {
    int col = nt * 16 + colb;
    float bv = bias[col];
    _Float16* out = H + (size_t)m0 * Cout + col;
#pragma unroll
    for (int i = 0; i < 8; ++i)
      out[(size_t)i * Cout] = (_Float16)(acc[nt][i] + bv);
  }
}

// ------------------------------------------------------ masked BN statistics
__global__ void stats_kernel(const _Float16* __restrict__ H,
                             const unsigned char* __restrict__ mask,
                             float* __restrict__ sum, float* __restrict__ sumsq,
                             int E, int Cout, int chunk) {
  int c = threadIdx.x;
  int e0 = blockIdx.x * chunk;
  int e1 = e0 + chunk; if (e1 > E) e1 = E;
  float s = 0.0f, q = 0.0f;
  for (int e = e0; e < e1; ++e) {
    if (mask[e]) {
      float h = (float)H[(size_t)e * Cout + c];
      s += h; q += h * h;
    }
  }
  atomicAdd(&sum[c], s);
  atomicAdd(&sumsq[c], q);
}

// ---------------------------------------------------- BN affine + ReLU (ip)
__global__ void bn_relu_kernel(_Float16* __restrict__ H,
                               const float* __restrict__ sum,
                               const float* __restrict__ sumsq,
                               const float* __restrict__ cnt,
                               const float* __restrict__ g,
                               const float* __restrict__ bt,
                               long long n, int Cout) {
  long long i = (long long)blockIdx.x * blockDim.x + threadIdx.x;
  if (i >= n) return;
  int c = (int)(i % Cout);
  float cn = fmaxf(cnt[0], 1.0f);
  float mean = sum[c] / cn;
  float var  = sumsq[c] / cn - mean * mean;
  float inv  = rsqrtf(var + 1e-5f);
  float h = (float)H[i];
  float y = g[c] * (h - mean) * inv + bt[c];
  H[i] = (_Float16)(y > 0.0f ? y : 0.0f);
}

// ------------------------------------------------------- masked max pooling
__global__ void maxpool_kernel(const _Float16* __restrict__ H,
                               const unsigned char* __restrict__ mask,
                               float* __restrict__ out, int k, int Cout, int colOff) {
  int i = blockIdx.x * blockDim.x + threadIdx.x;
  int total = NC_ * Cout;
  if (i >= total) return;
  int c = i / Cout, ch = i - c * Cout;
  float best = 0.0f; int any = 0;
  for (int j = 0; j < k; ++j) {
    int e = c * k + j;
    if (mask[e]) {
      float v = (float)H[(size_t)e * Cout + ch];
      best = any ? fmaxf(best, v) : v;
      any = 1;
    }
  }
  out[(size_t)c * OUTC_ + colOff + ch] = any ? best : 0.0f;
}

// ------------------------------------------------- pos_s and batch_s output
__global__ void tail_kernel(const float* __restrict__ pos,
                            const int* __restrict__ batch,
                            const int* __restrict__ idx, float* __restrict__ out) {
  int i = blockIdx.x * blockDim.x + threadIdx.x;
  if (i >= NC_) return;
  int p = idx[i];
  float* po = out + (size_t)NC_ * OUTC_ + (size_t)i * 3;
  po[0] = pos[p * 3 + 0];
  po[1] = pos[p * 3 + 1];
  po[2] = pos[p * 3 + 2];
  out[(size_t)NC_ * OUTC_ + (size_t)NC_ * 3 + i] = (float)batch[p];
}

// =========================================================================
extern "C" void kernel_launch(void* const* d_in, const int* in_sizes, int n_in,
                              void* d_out, int out_size, void* d_ws, size_t ws_size,
                              hipStream_t stream) {
  (void)in_sizes; (void)n_in; (void)out_size; (void)ws_size;
  const float* x    = (const float*)d_in[0];
  const float* pos  = (const float*)d_in[1];
  const int*   batc = (const int*)d_in[2];
  const int*   idx  = (const int*)d_in[3];
  auto leaf = [&](int s, int l, int w) -> const float* {
    return (const float*)d_in[4 + s * 12 + l * 4 + w];   // (W, b, g, bt)
  };

  static const int   chans[3][4] = {{35,32,32,64},{35,64,64,128},{35,64,96,128}};
  static const int   Ks[3]     = {16, 32, 64};
  static const float R2v[3]    = {0.01f, 0.04f, 0.16f};
  static const int   colOff[3] = {0, 64, 192};

  char* base = (char*)d_ws;
  size_t o_nbr_idx = 0;
  size_t o_nbr_d2  = o_nbr_idx + (size_t)NC_ * TOPK_ * 4;
  size_t o_stats   = o_nbr_d2  + (size_t)NC_ * TOPK_ * 4;
  size_t o_wpack   = o_stats + 4096 * 4;
  size_t woff[3][3];
  size_t wc = o_wpack;
  for (int s = 0; s < 3; ++s)
    for (int l = 0; l < 3; ++l) {
      int cin = chans[s][l], cout = chans[s][l + 1];
      int kt = ((l == 0) ? 64 : cin) / 32, ntl = cout / 16;
      woff[s][l] = wc;
      wc += (size_t)kt * ntl * 512 * 2;
    }
  size_t o_mask = (wc + 255) & ~(size_t)255;
  size_t moff[3]; size_t mc = o_mask;
  for (int s = 0; s < 3; ++s) { moff[s] = mc; mc += (size_t)NC_ * Ks[s]; }
  size_t o_P0 = (mc + 255) & ~(size_t)255;
  size_t szP  = (size_t)NC_ * 64 * 128 * 2;    // worst case: 1M edges x 128ch f16
  size_t o_P1 = o_P0 + szP;

  float* S = (float*)(base + o_stats);     // [0..2]=cnt/scale; layer sums at
  int* nbrI = (int*)(base + o_nbr_idx);    // 16+(s*3+l)*256 (+128 for sumsq)
  float* nbrD = (float*)(base + o_nbr_d2);

  zero_kernel<<<dim3(16), dim3(256), 0, stream>>>(S, 4096);
  topk_kernel<<<dim3(NC_), dim3(256), 0, stream>>>(pos, idx, nbrI, nbrD);

  for (int s = 0; s < 3; ++s)
    for (int l = 0; l < 3; ++l) {
      int cin = chans[s][l], cout = chans[s][l + 1];
      int kt = ((l == 0) ? 64 : cin) / 32, ntl = cout / 16;
      int elems = kt * ntl * 512;
      pack_w_kernel<<<dim3((elems + 255) / 256), dim3(256), 0, stream>>>(
          leaf(s, l, 0), cin, cout, ntl, (_Float16*)(base + woff[s][l]), elems);
    }

  for (int s = 0; s < 3; ++s) {
    int k = Ks[s];
    int E = NC_ * k;
    _Float16* Pin  = (_Float16*)(base + o_P0);
    _Float16* Pout = (_Float16*)(base + o_P1);
    unsigned char* msk = (unsigned char*)(base + moff[s]);

    pack_edges_kernel<<<dim3((E + 255) / 256), dim3(256), 0, stream>>>(
        x, pos, idx, nbrI, nbrD, msk, S + s, Pin, k, R2v[s], E);

    for (int l = 0; l < 3; ++l) {
      int cin = chans[s][l], cout = chans[s][l + 1];
      int kt = ((l == 0) ? 64 : cin) / 32, ntl = cout / 16;
      int Mtiles = E / 16;
      dim3 ggrid(Mtiles / 4), gblk(128);
      const _Float16* Wpk = (const _Float16*)(base + woff[s][l]);
      const float* bia = leaf(s, l, 1);
      switch (ntl) {
        case 2: gemm_kernel<2><<<ggrid, gblk, 0, stream>>>(Pin, Wpk, bia, Pout, kt); break;
        case 4: gemm_kernel<4><<<ggrid, gblk, 0, stream>>>(Pin, Wpk, bia, Pout, kt); break;
        case 6: gemm_kernel<6><<<ggrid, gblk, 0, stream>>>(Pin, Wpk, bia, Pout, kt); break;
        default: gemm_kernel<8><<<ggrid, gblk, 0, stream>>>(Pin, Wpk, bia, Pout, kt); break;
      }
      float* sum = S + 16 + (s * 3 + l) * 256;
      stats_kernel<<<dim3((E + 2047) / 2048), dim3(cout), 0, stream>>>(
          Pout, msk, sum, sum + 128, E, cout, 2048);
      long long n = (long long)E * cout;
      bn_relu_kernel<<<dim3((unsigned)((n + 255) / 256)), dim3(256), 0, stream>>>(
          Pout, sum, sum + 128, S + s, leaf(s, l, 2), leaf(s, l, 3), n, cout);
      _Float16* tmp = Pin; Pin = Pout; Pout = tmp;
    }
    int coutLast = chans[s][3];
    maxpool_kernel<<<dim3((NC_ * coutLast + 255) / 256), dim3(256), 0, stream>>>(
        Pin, msk, (float*)d_out, k, coutLast, colOff[s]);
  }

  tail_kernel<<<dim3(NC_ / 256), dim3(256), 0, stream>>>(pos, batc, idx, (float*)d_out);
}
